// VGN_31559419691756
// MI455X (gfx1250) — matrix operand order repaired and checked by reading, not compiled, not run on hardware
//
#include <hip/hip_runtime.h>
#include <stddef.h>

// ---------------------------------------------------------------------------
// GAT layer for MI455X (gfx1250, wave32, WMMA).
//   Phase 1: z = x @ W via v_wmma_f32_16x16x32_bf16; B fragments staged in LDS
//   Phase 2: per-vertex score scalars s_src/s_dst (factorized attention)
//   Phase 3: segment softmax (atomic max/add, L2-resident) + weighted aggregate
//   Phase 4: ELU in place on output
// ---------------------------------------------------------------------------

typedef __attribute__((ext_vector_type(16))) __bf16 v16bf;
typedef __attribute__((ext_vector_type(8)))  float  v8f;

#define IN_CH 128
#define HC    64          // H*C
#define NHEAD 4
#define CPH   16
#define SLOPE 0.2f

__device__ __forceinline__ __bf16 f2bf(float f) { return (__bf16)f; }

// ---------------- Phase 1: GEMM z[V,64] = x[V,128] @ W[128,64] -------------
// 256 threads = 8 waves; each wave owns a 16-row x 64-col tile.
// All waves share the same 16 B fragments -> build them once in LDS.
__global__ void __launch_bounds__(256) gat_gemm(const float* __restrict__ x,
                                                const float* __restrict__ W,
                                                float* __restrict__ z,
                                                int V) {
  __shared__ v16bf Bf[16][32];     // [ks*4+nt][lane], 16 KB

  // Cooperative fill: 512 fragment rows, 2 per thread.
  for (int fr = threadIdx.x; fr < 512; fr += 256) {
    const int fl   = fr & 31;        // lane slot
    const int frag = fr >> 5;        // ks*4+nt
    const int ks   = frag >> 2;
    const int nt   = frag & 3;
    const int fhi  = fl >> 4;
    const int fl16 = fl & 15;
    // B fragment (32x16 bf16): lane holds col N, elements j<8 -> K = k0+hi*8+j, j>=8 -> K = k0+16+hi*8+(j-8)
    const float* __restrict__ wp =
        W + (size_t)(ks * 32 + fhi * 8) * HC + nt * 16 + fl16;
    v16bf b;
#pragma unroll
    for (int j = 0; j < 16; ++j) b[j] = f2bf(wp[(size_t)(j < 8 ? j : j + 8) * HC]);
    Bf[frag][fl] = b;
  }
  __syncthreads();

  const int lane   = threadIdx.x & 31;
  const int wave   = threadIdx.x >> 5;
  const int lane16 = lane & 15;
  const int hi     = lane >> 4;        // 0: lanes 0-15, 1: lanes 16-31
  const int row0   = blockIdx.x * 128 + wave * 16;

  int arow = row0 + lane16;
  if (arow >= V) arow = V - 1;         // clamp: keep EXEC all-ones for WMMA
  const float* __restrict__ xrow = x + (size_t)arow * IN_CH;

  v8f acc[4];
#pragma unroll
  for (int nt = 0; nt < 4; ++nt)
#pragma unroll
    for (int r = 0; r < 8; ++r) acc[nt][r] = 0.0f;

#pragma unroll
  for (int ks = 0; ks < 4; ++ks) {
    const int k0 = ks * 32;
    // A fragment (16x32 bf16): lane holds row M=lane16.
    // elements 0..7  -> K = k0 + hi*8 + j
    // elements 8..15 -> K = k0 + 16 + hi*8 + (j-8)
    const float4 a0 = *(const float4*)(xrow + k0 + hi * 8);
    const float4 a1 = *(const float4*)(xrow + k0 + hi * 8 + 4);
    const float4 a2 = *(const float4*)(xrow + k0 + 16 + hi * 8);
    const float4 a3 = *(const float4*)(xrow + k0 + 16 + hi * 8 + 4);
    v16bf a;
    a[0]  = f2bf(a0.x); a[1]  = f2bf(a0.y); a[2]  = f2bf(a0.z); a[3]  = f2bf(a0.w);
    a[4]  = f2bf(a1.x); a[5]  = f2bf(a1.y); a[6]  = f2bf(a1.z); a[7]  = f2bf(a1.w);
    a[8]  = f2bf(a2.x); a[9]  = f2bf(a2.y); a[10] = f2bf(a2.z); a[11] = f2bf(a2.w);
    a[12] = f2bf(a3.x); a[13] = f2bf(a3.y); a[14] = f2bf(a3.z); a[15] = f2bf(a3.w);

#pragma unroll
    for (int nt = 0; nt < 4; ++nt) {
      const v16bf b = Bf[ks * 4 + nt][lane];   // 2x ds_load_b128
      acc[nt] = __builtin_amdgcn_wmma_f32_16x16x32_bf16(
          false, a, false, b, (short)0, acc[nt], false, false);
    }
  }

  // C/D layout: VGPR r -> M = hi*8 + r, N = lane16
#pragma unroll
  for (int nt = 0; nt < 4; ++nt) {
#pragma unroll
    for (int r = 0; r < 8; ++r) {
      const int row = row0 + hi * 8 + r;
      if (row < V) z[(size_t)row * HC + nt * 16 + lane16] = acc[nt][r];
    }
  }
}

// ---------------- Phase 2: per-vertex score scalars ------------------------
__global__ void __launch_bounds__(256) gat_scores(const float* __restrict__ z,
                                                  const float* __restrict__ a_src,
                                                  const float* __restrict__ a_dst,
                                                  float* __restrict__ s_src,
                                                  float* __restrict__ s_dst,
                                                  int V) {
  const int t = blockIdx.x * 256 + threadIdx.x;
  if (t >= V * NHEAD) return;
  const int v = t >> 2;
  const int h = t & 3;
  const float* __restrict__ zp = z + (size_t)v * HC + h * CPH;
  const float* __restrict__ as = a_src + h * CPH;
  const float* __restrict__ ad = a_dst + h * CPH;
  float ss = 0.0f, sd = 0.0f;
#pragma unroll
  for (int c = 0; c < CPH; ++c) {
    const float zv = zp[c];
    ss += zv * as[c];
    sd += zv * ad[c];
  }
  s_src[t] = ss;
  s_dst[t] = sd;
}

// ---------------- init: out=0, m=-1e9, denom=0 -----------------------------
__global__ void __launch_bounds__(256) gat_init(float* __restrict__ out,
                                                float* __restrict__ m,
                                                float* __restrict__ den,
                                                int V) {
  const int t = blockIdx.x * 256 + threadIdx.x;
  if (t < V * HC) out[t] = 0.0f;
  if (t < V * NHEAD) { m[t] = -1e9f; den[t] = 0.0f; }
}

// ---------------- Phase 3a: logits + segment max ---------------------------
// Stores the LeakyReLU'd logit into expe[t] so the exp pass needs no gathers.
__global__ void __launch_bounds__(256) gat_edge_max(const int* __restrict__ ei,
                                                    const float* __restrict__ s_src,
                                                    const float* __restrict__ s_dst,
                                                    float* __restrict__ expe,
                                                    float* __restrict__ m,
                                                    int E) {
  const int t = blockIdx.x * 256 + threadIdx.x;
  if (t >= E * NHEAD) return;
  const int e = t >> 2;
  const int h = t & 3;
  const int s = ei[e];
  const int d = ei[E + e];
  float v = s_src[s * NHEAD + h] + s_dst[d * NHEAD + h];
  v = (v > 0.0f) ? v : SLOPE * v;   // LeakyReLU
  expe[t] = v;
  atomicMax(m + d * NHEAD + h, v);  // global_atomic_max_num_f32
}

// ---------------- Phase 3b: exp + segment sum ------------------------------
__global__ void __launch_bounds__(256) gat_edge_exp(const int* __restrict__ ei,
                                                    const float* __restrict__ m,
                                                    float* __restrict__ expe,
                                                    float* __restrict__ den,
                                                    int E) {
  const int t = blockIdx.x * 256 + threadIdx.x;
  if (t >= E * NHEAD) return;
  const int e = t >> 2;
  const int h = t & 3;
  const int d = ei[E + e];
  const float ee = __expf(expe[t] - m[d * NHEAD + h]);
  expe[t] = ee;
  atomicAdd(den + d * NHEAD + h, ee);
}

// ---------------- Phase 3c: weighted aggregate -----------------------------
// one thread per (edge, head): alpha amortized over 16 channels,
// gathers as float4; z/out/denom are L2-resident (192MB L2).
__global__ void __launch_bounds__(256) gat_agg(const int* __restrict__ ei,
                                               const float* __restrict__ z,
                                               const float* __restrict__ expe,
                                               const float* __restrict__ den,
                                               float* __restrict__ out,
                                               int E) {
  const int t = blockIdx.x * 256 + threadIdx.x;
  if (t >= E * NHEAD) return;
  const int e = t >> 2;
  const int h = t & 3;
  const int s = ei[e];
  const int d = ei[E + e];
  const float alpha = expe[t] / (den[d * NHEAD + h] + 1e-9f);
  const float4* __restrict__ zp =
      (const float4*)(z + (size_t)s * HC + h * CPH);
  float* __restrict__ op = out + (size_t)d * HC + h * CPH;
#pragma unroll
  for (int q = 0; q < 4; ++q) {
    const float4 zv = zp[q];
    atomicAdd(op + q * 4 + 0, zv.x * alpha);
    atomicAdd(op + q * 4 + 1, zv.y * alpha);
    atomicAdd(op + q * 4 + 2, zv.z * alpha);
    atomicAdd(op + q * 4 + 3, zv.w * alpha);
  }
}

// ---------------- Phase 4: ELU in place ------------------------------------
__global__ void __launch_bounds__(256) gat_elu(float* __restrict__ out, int V) {
  const int t = blockIdx.x * 256 + threadIdx.x;
  if (t >= V * HC) return;
  const float v = out[t];
  out[t] = (v > 0.0f) ? v : (__expf(v) - 1.0f);
}

// ---------------------------------------------------------------------------
extern "C" void kernel_launch(void* const* d_in, const int* in_sizes, int n_in,
                              void* d_out, int out_size, void* d_ws, size_t ws_size,
                              hipStream_t stream) {
  const float* x      = (const float*)d_in[0];   // [V,128]
  const int*   ei     = (const int*)  d_in[1];   // [2,E]
  const float* W      = (const float*)d_in[2];   // [128,64]
  const float* a_src  = (const float*)d_in[3];   // [4,16]
  const float* a_dst  = (const float*)d_in[4];   // [4,16]
  float*       out    = (float*)d_out;           // [V,64]

  const int V = in_sizes[0] / IN_CH;
  const int E = in_sizes[1] / 2;

  // workspace layout (floats)
  float* ws     = (float*)d_ws;
  float* z      = ws;                                // V*64
  float* s_src  = z     + (size_t)V * HC;            // V*4
  float* s_dst  = s_src + (size_t)V * NHEAD;         // V*4
  float* m      = s_dst + (size_t)V * NHEAD;         // V*4
  float* den    = m     + (size_t)V * NHEAD;         // V*4
  float* expe   = den   + (size_t)V * NHEAD;         // E*4

  const int gemm_blocks = (V + 127) / 128;
  const int vh_blocks   = (V * NHEAD + 255) / 256;
  const int vhc_blocks  = (V * HC + 255) / 256;
  const int eh_blocks   = (E * NHEAD + 255) / 256;

  gat_gemm    <<<gemm_blocks, 256, 0, stream>>>(x, W, z, V);
  gat_scores  <<<vh_blocks,   256, 0, stream>>>(z, a_src, a_dst, s_src, s_dst, V);
  gat_init    <<<vhc_blocks,  256, 0, stream>>>(out, m, den, V);
  gat_edge_max<<<eh_blocks,   256, 0, stream>>>(ei, s_src, s_dst, expe, m, E);
  gat_edge_exp<<<eh_blocks,   256, 0, stream>>>(ei, m, expe, den, E);
  gat_agg     <<<eh_blocks,   256, 0, stream>>>(ei, z, expe, den, out, E);
  gat_elu     <<<vhc_blocks,  256, 0, stream>>>(out, V);
}
